// MotifAttention_13005160972571
// MI455X (gfx1250) — matrix operand order repaired and checked by reading, not compiled, not run on hardware
//
#include <hip/hip_runtime.h>
#include <hip/hip_bf16.h>
#include <math.h>

// ---------------------------------------------------------------------------
// Types for CDNA5 WMMA (wave32, 16x16x32 bf16 -> f32)
// ---------------------------------------------------------------------------
typedef __attribute__((ext_vector_type(16))) __bf16 v16bf;
typedef __attribute__((ext_vector_type(8)))  float  v8f;

union BFrag { v16bf v; unsigned short u[16]; unsigned int w[8]; };
union FAcc  { v8f  v; float f[8]; };

__device__ __forceinline__ unsigned short f2bf(float x) {
  unsigned int u = __float_as_uint(x);
  u = (u + 0x7FFFu + ((u >> 16) & 1u)) >> 16;   // round-to-nearest-even
  return (unsigned short)u;
}

// CDNA5 async global->LDS copy (ASYNCcnt-tracked, bypasses VGPRs).
// GVS addressing: mem = SGPR64 base + 32-bit VGPR byte offset.
__device__ __forceinline__ void async_ld_b128(unsigned lds_byte_addr,
                                              const void* base,
                                              unsigned gbyte_off) {
  asm volatile("global_load_async_to_lds_b128 %0, %1, %2"
               :: "v"(lds_byte_addr), "v"(gbyte_off),
                  "s"((unsigned long long)(uintptr_t)base)
               : "memory");
}
template <int N>
__device__ __forceinline__ void wait_async_le() {
  asm volatile("s_wait_asynccnt %0" :: "i"(N) : "memory");
}
__device__ __forceinline__ unsigned lds_off(const void* p) {
  return (unsigned)(uintptr_t)p;   // low 32 bits of generic LDS ptr = offset
}

#define TSEQ 2048
#define HDIM 2048
#define NQH  16
#define NKVH 8
#define DH   128
#define ATTN_SCALE 0.08838834764831845f  // 1/sqrt(128)

// ---------------------------------------------------------------------------
// fp32 -> bf16 bulk convert
// ---------------------------------------------------------------------------
__global__ __launch_bounds__(256) void f32_to_bf16_kernel(
    const float* __restrict__ in, unsigned short* __restrict__ out, unsigned n) {
  unsigned i = blockIdx.x * 256u + threadIdx.x;
  if (i < n) out[i] = f2bf(in[i]);
}

// ---------------------------------------------------------------------------
// bf16 x bf16 -> f32 GEMM, 128x128 tile, BK=32, 256 threads.
// 8 waves in a 4x2 grid; each wave owns 32x64 = 2x4 WMMA tiles (8 WMMA/step).
// Double-buffered software pipeline:
//   - A tile k+1 DMA'd via global_load_async_to_lds_b128 while tile k computes
//     (async loads retire in order -> s_wait_asynccnt 2 isolates tile k)
//   - B tile k+1 global loads prefetched to VGPRs; only the LDS transpose
//     scatter ([n][k] layout for contiguous ds_load b32/b128 frags) is sync.
// ---------------------------------------------------------------------------
__global__ __launch_bounds__(256) void gemm_bf16_128x128(
    const unsigned short* __restrict__ A,
    const unsigned short* __restrict__ B,
    float* __restrict__ Cf,
    int M, int N, int K) {
  __shared__ unsigned short As[2][128][40];  // [buf][m][k], pad->16B rows
  __shared__ unsigned short Bs[2][128][40];  // [buf][n][k] (transposed)

  const int tid  = threadIdx.x;
  const int lane = tid & 31, wave = tid >> 5;
  const int lm = lane & 15, lh = lane >> 4;
  const int bm = blockIdx.y * 128, bn = blockIdx.x * 128;
  const int wr = (wave & 3) * 32, wc = (wave >> 2) * 64;

  // Per-thread staging coordinates (2 chunks of 8 bf16 each)
  int ar[2], ac[2], br[2], bc[2];
#pragma unroll
  for (int i = 0; i < 2; ++i) {
    int chunk = i * 256 + tid;
    ar[i] = chunk >> 2;  ac[i] = (chunk & 3) << 3;    // A: 128 rows x 32 k
    br[i] = chunk >> 4;  bc[i] = (chunk & 15) << 3;   // B: 32 k x 128 n
  }

  FAcc acc[2][4];
#pragma unroll
  for (int i = 0; i < 2; ++i)
#pragma unroll
    for (int j = 0; j < 4; ++j)
#pragma unroll
      for (int e = 0; e < 8; ++e) acc[i][j].f[e] = 0.0f;

  const int nk = K >> 5;
  uint4 bpref[2];

  // Prologue: tile 0 -> buffer 0 (A async, B to registers)
#pragma unroll
  for (int i = 0; i < 2; ++i)
    async_ld_b128(lds_off(&As[0][ar[i]][ac[i]]), A,
                  (unsigned)(((bm + ar[i]) * K + ac[i]) * 2));
#pragma unroll
  for (int i = 0; i < 2; ++i)
    bpref[i] = *(const uint4*)(B + (unsigned)(br[i] * N + bn + bc[i]));

  for (int kt = 0; kt < nk; ++kt) {
    const int cur = kt & 1;
    // Scatter prefetched B tile kt into Bs[cur] (transpose)
#pragma unroll
    for (int i = 0; i < 2; ++i) {
      int r = br[i], c = bc[i];
      uint4 d = bpref[i];
      Bs[cur][c + 0][r] = (unsigned short)(d.x & 0xffff);
      Bs[cur][c + 1][r] = (unsigned short)(d.x >> 16);
      Bs[cur][c + 2][r] = (unsigned short)(d.y & 0xffff);
      Bs[cur][c + 3][r] = (unsigned short)(d.y >> 16);
      Bs[cur][c + 4][r] = (unsigned short)(d.z & 0xffff);
      Bs[cur][c + 5][r] = (unsigned short)(d.z >> 16);
      Bs[cur][c + 6][r] = (unsigned short)(d.w & 0xffff);
      Bs[cur][c + 7][r] = (unsigned short)(d.w >> 16);
    }
    // Prefetch tile kt+1 (overlaps with this tile's WMMAs)
    if (kt + 1 < nk) {
      const int k0n = (kt + 1) << 5;
#pragma unroll
      for (int i = 0; i < 2; ++i)
        async_ld_b128(lds_off(&As[cur ^ 1][ar[i]][ac[i]]), A,
                      (unsigned)(((bm + ar[i]) * K + k0n + ac[i]) * 2));
#pragma unroll
      for (int i = 0; i < 2; ++i)
        bpref[i] = *(const uint4*)(B + (unsigned)((k0n + br[i]) * N + bn + bc[i]));
      wait_async_le<2>();   // tile kt's DMA done (in-order retire)
    } else {
      wait_async_le<0>();
    }
    __syncthreads();        // tile kt fully visible in LDS

    // A fragments: 16x32, lane lm = row, vgpr v -> k = {2v | 16+2(v-4)} + 8*lh
    BFrag af[2], bf[4];
#pragma unroll
    for (int i = 0; i < 2; ++i) {
      int m = wr + i * 16 + lm;
#pragma unroll
      for (int v = 0; v < 8; ++v) {
        int kk = ((v < 4) ? (2 * v) : (16 + 2 * (v - 4))) + lh * 8;
        af[i].w[v] = *(const unsigned int*)&As[cur][m][kk];
      }
    }
    // B fragments: 32x16, lane lm = col n, vgpr v -> k = {2v | 16+2(v-4)} + 8*lh
#pragma unroll
    for (int j = 0; j < 4; ++j) {
      int n = wc + j * 16 + lm;
#pragma unroll
      for (int v = 0; v < 8; ++v) {
        int kk = ((v < 4) ? (2 * v) : (16 + 2 * (v - 4))) + lh * 8;
        bf[j].w[v] = *(const unsigned int*)&Bs[cur][n][kk];
      }
    }
#pragma unroll
    for (int i = 0; i < 2; ++i)
#pragma unroll
      for (int j = 0; j < 4; ++j)
        acc[i][j].v = __builtin_amdgcn_wmma_f32_16x16x32_bf16(
            false, af[i].v, false, bf[j].v, (short)0, acc[i][j].v, false, false);
    __syncthreads();        // all waves done reading buf cur -> reusable
  }

  // C layout: vgpr e -> row e + 8*lh, col lm. 32-bit offsets -> saddr form.
#pragma unroll
  for (int i = 0; i < 2; ++i)
#pragma unroll
    for (int j = 0; j < 4; ++j)
#pragma unroll
      for (int e = 0; e < 8; ++e) {
        unsigned row = (unsigned)(bm + wr + i * 16 + e + lh * 8);
        unsigned col = (unsigned)(bn + wc + j * 16 + lm);
        Cf[row * (unsigned)N + col] = acc[i][j].f[e];
      }
}

// ---------------------------------------------------------------------------
// RoPE (NeoX) on q/k heads of the fp32 qkv buffer; emit bf16 Q[T][16][128],
// K[T][8][128], V[T][8][128].
// ---------------------------------------------------------------------------
__global__ __launch_bounds__(256) void rope_split_kernel(
    const float* __restrict__ qkv, const int* __restrict__ pos,
    unsigned short* __restrict__ Qo, unsigned short* __restrict__ Ko,
    unsigned short* __restrict__ Vo) {
  const int t = blockIdx.x;
  const float fp = (float)pos[t];
  const float* row = qkv + (unsigned)t * 4096u;

  for (int i = threadIdx.x; i < 24 * 64; i += blockDim.x) {
    int hd = i >> 6, j = i & 63;
    float inv = __expf(-(float)j * (9.210340371976184f / 64.0f));  // ln(1e4)/64
    float ang = fp * inv;
    float s, c;
    __sincosf(ang, &s, &c);
    if (hd < NQH) {
      const float* b = row + hd * DH;
      float x1 = b[j], x2 = b[j + 64];
      unsigned short* ob = Qo + ((unsigned)t * NQH + hd) * DH;
      ob[j]      = f2bf(x1 * c - x2 * s);
      ob[j + 64] = f2bf(x2 * c + x1 * s);
    } else {
      int kh = hd - NQH;
      const float* b = row + 2048 + kh * DH;
      float x1 = b[j], x2 = b[j + 64];
      unsigned short* ob = Ko + ((unsigned)t * NKVH + kh) * DH;
      ob[j]      = f2bf(x1 * c - x2 * s);
      ob[j + 64] = f2bf(x2 * c + x1 * s);
    }
  }
  for (int i = threadIdx.x; i < NKVH * DH; i += blockDim.x)
    Vo[(unsigned)t * (NKVH * DH) + i] = f2bf(row[3072 + i]);
}

// ---------------------------------------------------------------------------
// lambda = exp(sum lq1*lk1) - exp(sum lq2*lk2) + 0.2   (one wave)
// ---------------------------------------------------------------------------
__global__ __launch_bounds__(32) void lambda_kernel(
    const float* __restrict__ lq1, const float* __restrict__ lk1,
    const float* __restrict__ lq2, const float* __restrict__ lk2,
    float* __restrict__ lam) {
  int lane = threadIdx.x;
  float s1 = 0.0f, s2 = 0.0f;
  for (int i = lane; i < DH; i += 32) {
    s1 += lq1[i] * lk1[i];
    s2 += lq2[i] * lk2[i];
  }
#pragma unroll
  for (int off = 16; off > 0; off >>= 1) {
    s1 += __shfl_xor(s1, off, 32);
    s2 += __shfl_xor(s2, off, 32);
  }
  if (lane == 0) *lam = __expf(s1) - __expf(s2) + 0.2f;
}

// ---------------------------------------------------------------------------
// Flash attention, 16 problems = (branch in {0,1}) x (pair in 0..7).
//   qhead = 2*pair+branch, khead = 2*(pair/2)+branch,
//   V = [v1 | v2] heads 2*(pair/2), 2*(pair/2)+1  -> 256-dim output.
// Block: 64 queries, 4 waves x 16 rows. 32 keys per step (2 WMMA score tiles).
// K staged via async LDS DMA; V staged transposed for contiguous PV B-frags.
// ---------------------------------------------------------------------------
__global__ __launch_bounds__(128) void attn_diff_kernel(
    const unsigned short* __restrict__ Q,   // [T][16][128]
    const unsigned short* __restrict__ Kc,  // [T][8][128]
    const unsigned short* __restrict__ Vc,  // [T][8][128]
    float* __restrict__ Out) {              // [2][T][8][256]
  __shared__ unsigned short Ks[32][136];    // [key][chan]
  __shared__ unsigned short Vs[256][36];    // transposed [dim][key]
  __shared__ unsigned short Ps[4][16][40];  // per-wave P repack buffer

  const int tid  = threadIdx.x;
  const int lane = tid & 31, wave = tid >> 5;
  const int lm = lane & 15, lh = lane >> 4;
  const int prob = blockIdx.y;
  const int branch = prob >> 3, pair = prob & 7;
  const int qhead = 2 * pair + branch;
  const int khead = 2 * (pair >> 1) + branch;
  const int v1h = 2 * (pair >> 1), v2h = v1h + 1;
  const int qblk = blockIdx.x * 64;
  const int q0 = qblk + wave * 16;

  // Persistent Q fragments (4 chunks of 32 channels)
  BFrag qf[4];
  {
    const unsigned short* qp = Q + ((unsigned)(q0 + lm) * NQH + qhead) * DH;
#pragma unroll
    for (int c = 0; c < 4; ++c)
#pragma unroll
      for (int v = 0; v < 8; ++v) {
        int ch = c * 32 + ((v < 4) ? 2 * v : 16 + 2 * (v - 4)) + lh * 8;
        qf[c].w[v] = *(const unsigned int*)(qp + ch);
      }
  }

  FAcc o[16];                                // 16 out tiles -> 256 dims
#pragma unroll
  for (int t = 0; t < 16; ++t)
#pragma unroll
    for (int e = 0; e < 8; ++e) o[t].f[e] = 0.0f;
  float mrow[8], lrow[8];
#pragma unroll
  for (int e = 0; e < 8; ++e) { mrow[e] = -3.0e38f; lrow[e] = 0.0f; }

  const int nsteps = qblk / 32 + 2;          // uniform across block (causal)
  for (int s = 0; s < nsteps; ++s) {
    const int kk0 = s * 32;
    // Stage K tile 32x128 via async LDS DMA
#pragma unroll
    for (int i = 0; i < 4; ++i) {
      int chunk = i * 128 + tid;
      int key = chunk >> 4, c0 = (chunk & 15) << 3;
      async_ld_b128(lds_off(&Ks[key][c0]), Kc,
                    (unsigned)((((kk0 + key) * NKVH + khead) * DH + c0) * 2));
    }
    // Stage V tile 32x256 transposed ([dim][key]); dims 0..127=v1, 128..255=v2
#pragma unroll
    for (int i = 0; i < 8; ++i) {
      int chunk = i * 128 + tid;
      int key = chunk >> 5, d0 = (chunk & 31) << 3;
      int vh = (d0 < 128) ? v1h : v2h;
      uint4 d = *(const uint4*)(Vc +
          (unsigned)(((kk0 + key) * NKVH + vh) * DH + (d0 & 127)));
      Vs[d0 + 0][key] = (unsigned short)(d.x & 0xffff);
      Vs[d0 + 1][key] = (unsigned short)(d.x >> 16);
      Vs[d0 + 2][key] = (unsigned short)(d.y & 0xffff);
      Vs[d0 + 3][key] = (unsigned short)(d.y >> 16);
      Vs[d0 + 4][key] = (unsigned short)(d.z & 0xffff);
      Vs[d0 + 5][key] = (unsigned short)(d.z >> 16);
      Vs[d0 + 6][key] = (unsigned short)(d.w & 0xffff);
      Vs[d0 + 7][key] = (unsigned short)(d.w >> 16);
    }
    wait_async_le<0>();
    __syncthreads();

    // Scores: two 16x16 tiles, each 4 chained WMMAs over the 128 channels
    FAcc sc[2];
#pragma unroll
    for (int tt = 0; tt < 2; ++tt) {
#pragma unroll
      for (int e = 0; e < 8; ++e) sc[tt].f[e] = 0.0f;
      int keyn = tt * 16 + lm;
#pragma unroll
      for (int c = 0; c < 4; ++c) {
        BFrag kf;
#pragma unroll
        for (int v = 0; v < 8; ++v) {
          int ch = c * 32 + ((v < 4) ? 2 * v : 16 + 2 * (v - 4)) + lh * 8;
          kf.w[v] = *(const unsigned int*)&Ks[keyn][ch];
        }
        sc[tt].v = __builtin_amdgcn_wmma_f32_16x16x32_bf16(
            false, qf[c].v, false, kf.v, (short)0, sc[tt].v, false, false);
      }
    }
    // Scale + causal mask
#pragma unroll
    for (int tt = 0; tt < 2; ++tt)
#pragma unroll
      for (int e = 0; e < 8; ++e) {
        int qr = q0 + e + lh * 8;
        int ky = kk0 + tt * 16 + lm;
        float v = sc[tt].f[e] * ATTN_SCALE;
        sc[tt].f[e] = (ky <= qr) ? v : -1.0e30f;
      }
    // Online softmax: row reductions across the 16-lane half (wave32)
    float pscale[8];
#pragma unroll
    for (int e = 0; e < 8; ++e) {
      float rm = fmaxf(sc[0].f[e], sc[1].f[e]);
#pragma unroll
      for (int off = 1; off < 16; off <<= 1)
        rm = fmaxf(rm, __shfl_xor(rm, off, 32));
      float mn = fmaxf(mrow[e], rm);
      pscale[e] = __expf(mrow[e] - mn);
      mrow[e] = mn;
      float p0 = __expf(sc[0].f[e] - mn);
      float p1 = __expf(sc[1].f[e] - mn);
      sc[0].f[e] = p0; sc[1].f[e] = p1;
      float rs = p0 + p1;
#pragma unroll
      for (int off = 1; off < 16; off <<= 1)
        rs += __shfl_xor(rs, off, 32);
      lrow[e] = lrow[e] * pscale[e] + rs;
    }
#pragma unroll
    for (int t = 0; t < 16; ++t)
#pragma unroll
      for (int e = 0; e < 8; ++e) o[t].f[e] *= pscale[e];

    // Repack P (C layout) -> A-fragment layout via per-wave LDS
#pragma unroll
    for (int tt = 0; tt < 2; ++tt)
#pragma unroll
      for (int e = 0; e < 8; ++e)
        Ps[wave][e + lh * 8][tt * 16 + lm] = f2bf(sc[tt].f[e]);
    asm volatile("s_wait_dscnt 0x0" ::: "memory");
    BFrag pf;
#pragma unroll
    for (int v = 0; v < 8; ++v) {
      int kk = ((v < 4) ? 2 * v : 16 + 2 * (v - 4)) + lh * 8;
      pf.w[v] = *(const unsigned int*)&Ps[wave][lm][kk];
    }
    // P(16x32) x V(32x256): 16 WMMAs
#pragma unroll
    for (int t = 0; t < 16; ++t) {
      BFrag vf;
      int n = t * 16 + lm;
#pragma unroll
      for (int v = 0; v < 8; ++v) {
        int kk = ((v < 4) ? 2 * v : 16 + 2 * (v - 4)) + lh * 8;
        vf.w[v] = *(const unsigned int*)&Vs[n][kk];
      }
      o[t].v = __builtin_amdgcn_wmma_f32_16x16x32_bf16(
          false, pf.v, false, vf.v, (short)0, o[t].v, false, false);
    }
    __syncthreads();
  }

  float inv[8];
#pragma unroll
  for (int e = 0; e < 8; ++e) inv[e] = 1.0f / lrow[e];
#pragma unroll
  for (int t = 0; t < 16; ++t)
#pragma unroll
    for (int e = 0; e < 8; ++e) {
      unsigned qr  = (unsigned)(q0 + e + lh * 8);
      unsigned dim = (unsigned)(t * 16 + lm);
      Out[(((unsigned)branch * TSEQ + qr) * 8u + (unsigned)pair) * 256u + dim] =
          o[t].f[e] * inv[e];
    }
}

// ---------------------------------------------------------------------------
// a = a1 - lam*a2, RMSNorm over 256 dims, * subln_w * (1-0.2), -> bf16
// One block per (t, pair) row of 256 dims.
// ---------------------------------------------------------------------------
__global__ __launch_bounds__(256) void combine_kernel(
    const float* __restrict__ A12, const float* __restrict__ lamp,
    const float* __restrict__ subw, unsigned short* __restrict__ outbf) {
  __shared__ float red[256];
  const unsigned row = blockIdx.x;   // t*8 + pair
  const unsigned d = threadIdx.x;
  const float lam = *lamp;
  const unsigned idx = row * 256u + d;
  const float v = A12[idx] - lam * A12[(unsigned)TSEQ * 2048u + idx];
  red[d] = v * v;
  __syncthreads();
  for (int s = 128; s > 0; s >>= 1) {
    if ((int)d < s) red[d] += red[d + s];
    __syncthreads();
  }
  float r = rsqrtf(red[0] * (1.0f / 256.0f) + 1e-5f);
  outbf[idx] = f2bf(v * r * subw[d] * 0.8f);
}

// ---------------------------------------------------------------------------
extern "C" void kernel_launch(void* const* d_in, const int* in_sizes, int n_in,
                              void* d_out, int out_size, void* d_ws, size_t ws_size,
                              hipStream_t stream) {
  (void)in_sizes; (void)n_in; (void)out_size; (void)ws_size;
  const int*   pos  = (const int*)d_in[0];
  const float* hs   = (const float*)d_in[1];
  const float* Wqkv = (const float*)d_in[2];
  const float* Wo   = (const float*)d_in[3];
  const float* lq1  = (const float*)d_in[4];
  const float* lk1  = (const float*)d_in[5];
  const float* lq2  = (const float*)d_in[6];
  const float* lk2  = (const float*)d_in[7];
  const float* subw = (const float*)d_in[8];
  float* out = (float*)d_out;

  char* ws = (char*)d_ws;
  size_t off = 0;
  auto alloc = [&](size_t bytes) {
    char* p = ws + off;
    off += (bytes + 255) & ~(size_t)255;
    return (void*)p;
  };
  unsigned short* hs_bf   = (unsigned short*)alloc((size_t)TSEQ * HDIM * 2);
  unsigned short* wqkv_bf = (unsigned short*)alloc((size_t)HDIM * 4096 * 2);
  unsigned short* wo_bf   = (unsigned short*)alloc((size_t)HDIM * HDIM * 2);
  unsigned short* a_bf    = (unsigned short*)alloc((size_t)TSEQ * 2048 * 2);
  float*          qkv_f   = (float*)alloc((size_t)TSEQ * 4096 * 4);
  unsigned short* q_bf    = (unsigned short*)alloc((size_t)TSEQ * 2048 * 2);
  unsigned short* k_bf    = (unsigned short*)alloc((size_t)TSEQ * 1024 * 2);
  unsigned short* v_bf    = (unsigned short*)alloc((size_t)TSEQ * 1024 * 2);
  float*          a12     = (float*)alloc((size_t)2 * TSEQ * 2048 * 4);
  float*          lam     = (float*)alloc(256);

  // 1) bf16 conversions of GEMM operands
  f32_to_bf16_kernel<<<(TSEQ * HDIM) / 256, 256, 0, stream>>>(hs, hs_bf, TSEQ * HDIM);
  f32_to_bf16_kernel<<<(HDIM * 4096) / 256, 256, 0, stream>>>(Wqkv, wqkv_bf, HDIM * 4096);
  f32_to_bf16_kernel<<<(HDIM * HDIM) / 256, 256, 0, stream>>>(Wo, wo_bf, HDIM * HDIM);

  // 2) QKV projection: [2048,2048] x [2048,4096] -> fp32
  gemm_bf16_128x128<<<dim3(4096 / 128, TSEQ / 128), 256, 0, stream>>>(
      hs_bf, wqkv_bf, qkv_f, TSEQ, 4096, HDIM);

  // 3) RoPE + head split -> bf16 Q/K/V
  rope_split_kernel<<<TSEQ, 256, 0, stream>>>(qkv_f, pos, q_bf, k_bf, v_bf);

  // 4) lambda scalar
  lambda_kernel<<<1, 32, 0, stream>>>(lq1, lk1, lq2, lk2, lam);

  // 5) 16 causal attention problems, flash-style WMMA
  attn_diff_kernel<<<dim3(TSEQ / 64, 16), 128, 0, stream>>>(q_bf, k_bf, v_bf, a12);

  // 6) differential combine + RMSNorm -> bf16 activations
  combine_kernel<<<TSEQ * 8, 256, 0, stream>>>(a12, lam, subw, a_bf);

  // 7) output projection: [2048,2048] x [2048,2048] -> fp32 d_out
  gemm_bf16_128x128<<<dim3(HDIM / 128, TSEQ / 128), 256, 0, stream>>>(
      a_bf, wo_bf, out, TSEQ, HDIM, HDIM);
}
